// PointProp_1125281431922
// MI455X (gfx1250) — matrix operand order repaired and checked by reading, not compiled, not run on hardware
//
#include <hip/hip_runtime.h>
#include <hip/hip_bf16.h>

// ---------------------------------------------------------------------------
// PointProp fused pipeline for gfx1250 (MI455X), bf16 WMMA, f32 accumulate.
//   right = (sum_k comps) @ Wm^T + 8*bm ; right = right @ Wu^T + bu
//   x = [signal | right] ; 3x (Lin+ReLU) ; out = x @ W3^T + b3
// N=65536, D=E=256, H=132 (padded to 160), K=8 components.
// Each wave owns 32 rows (two 16-row M-tiles): every B fragment feeds 2 WMMAs.
// ---------------------------------------------------------------------------

#define NPTS  65536
#define DDIM  256
#define EDIM  256
#define HDIM  132
#define KCOMP 8

#define ROWS    64         // rows per block (2 waves x 32 rows)
#define THREADS 64         // 2 waves (wave32)
#define XW      516        // LDS row stride (halves) for [signal|right] (512 + 4 pad)
#define HW      164        // LDS row stride (halves) for hidden (160 + 4 pad)

typedef __attribute__((ext_vector_type(16))) __bf16 v16bf;
typedef __attribute__((ext_vector_type(8)))  float  v8f;
typedef __attribute__((ext_vector_type(4)))  float  f4;

union ABFrag { v16bf v; unsigned int u[8]; };

__device__ __forceinline__ unsigned short f2bf(float f) {
  unsigned int u = __float_as_uint(f);
  unsigned int r = u + 0x7FFFu + ((u >> 16) & 1u);   // round-to-nearest-even
  return (unsigned short)(r >> 16);
}
__device__ __forceinline__ unsigned int pack2(float a, float b) {
  return (unsigned int)f2bf(a) | ((unsigned int)f2bf(b) << 16);
}

// ---- prepped-weight fragment offsets in d_ws (units: halves / uint16) -----
#define OFF_WM  0u                         // 16x8  frags
#define OFF_WU  (OFF_WM + 16u*8u*512u)     // 16x8
#define OFF_W0  (OFF_WU + 16u*8u*512u)     // 10x16
#define OFF_W1  (OFF_W0 + 10u*16u*512u)    // 10x5
#define OFF_W2  (OFF_W1 + 10u*5u*512u)     // 10x5
#define OFF_W3  (OFF_W2 + 10u*5u*512u)     // 16x5

__global__ void prep_w(const float* __restrict__ W, unsigned int* __restrict__ dst,
                       int outDim, int inDim, int NT, int KT) {
  int idx = blockIdx.x * blockDim.x + threadIdx.x;
  int total = NT * KT * 256;               // dwords
  if (idx >= total) return;
  int frag   = idx >> 8;
  int within = idx & 255;
  int lane   = within >> 3;
  int j      = within & 7;
  int nt = frag / KT, kt = frag - nt * KT;
  int n  = nt * 16 + (lane & 15);
  int K  = kt * 32 + ((lane >> 4) << 4) + 2 * j;
  float v0 = (n < outDim && K     < inDim) ? W[(size_t)n * inDim + K]     : 0.f;
  float v1 = (n < outDim && K + 1 < inDim) ? W[(size_t)n * inDim + K + 1] : 0.f;
  dst[(size_t)frag * 256 + lane * 8 + j] = pack2(v0, v1);
}

// ---------------------------------------------------------------------------
__global__ __launch_bounds__(THREADS)
void fused_main(const float* __restrict__ signal,
                const float* __restrict__ comps,
                const float* __restrict__ bm, const float* __restrict__ bu,
                const float* __restrict__ b0, const float* __restrict__ b1,
                const float* __restrict__ b2, const float* __restrict__ b3,
                const unsigned short* __restrict__ wsW,
                float* __restrict__ out) {
  __shared__ unsigned short xbuf[ROWS * XW];   // [64][512+pad]: signal | right
  __shared__ unsigned short hbuf[ROWS * HW];   // [64][160+pad]: hidden acts

  const int tid     = threadIdx.x;
  const int rowbase = blockIdx.x * ROWS;

  // ------- Phase 1: stream components (536MB, non-temporal), sum, bf16 -----
  {
    const f4* sig4 = (const f4*)signal;
    const f4* cmp4 = (const f4*)comps;
    for (int i = tid; i < ROWS * (DDIM / 4); i += THREADS) {
      int r  = i >> 6;            // row in tile
      int c4 = i & 63;            // float4 column
      f4 s = __builtin_nontemporal_load(&sig4[(size_t)(rowbase + r) * (EDIM / 4) + c4]);
      *(uint2*)&xbuf[r * XW + c4 * 4] = make_uint2(pack2(s.x, s.y), pack2(s.z, s.w));
      f4 a = {0.f, 0.f, 0.f, 0.f};
#pragma unroll
      for (int k = 0; k < KCOMP; ++k) {
        f4 v = __builtin_nontemporal_load(
            &cmp4[((size_t)k * NPTS + rowbase + r) * (DDIM / 4) + c4]);
        a += v;
      }
      *(uint2*)&xbuf[r * XW + DDIM + c4 * 4] = make_uint2(pack2(a.x, a.y), pack2(a.z, a.w));
    }
  }
  __syncthreads();

  // ------- Phase 2: per-wave 32-row (2 M-tile) GEMM chain ------------------
  const int wave = tid >> 5;
  const int lane = tid & 31;
  const int m0a  = wave * 32;        // first M-tile
  const int m0b  = m0a + 16;         // second M-tile
  const int lm   = lane & 15;
  const int lg   = lane >> 4;

  // A fragment (16x32 bf16): lane<16 K in {0..15}, lane>=16 K in {16..31}
  auto loadA = [&](const unsigned short* base, int stride, int kt, int m0) {
    ABFrag a;
#pragma unroll
    for (int j = 0; j < 8; ++j) {
      int Koff = kt * 32 + 16 * lg + 2 * j;
      a.u[j] = *(const unsigned int*)&base[(m0 + lm) * stride + Koff];
    }
    return a;
  };
  auto loadB = [&](unsigned int off, int KT, int nt, int kt) {
    return *(const v16bf*)&wsW[off + ((size_t)(nt * KT + kt)) * 512 + lane * 16];
  };
  auto storeD = [&](unsigned short* base, int stride, int nt, v8f acc,
                    const float* bias, float bscale, bool relu, int valid, int m0) {
    int col = nt * 16 + lm;
    float b = bias[col < valid ? col : 0];
    b = (col < valid) ? bscale * b : 0.f;
#pragma unroll
    for (int r = 0; r < 8; ++r) {
      float v = acc[r] + b;
      if (relu) v = fmaxf(v, 0.f);
      base[(m0 + r + 8 * lg) * stride + col] = f2bf(v);
    }
  };
  auto zero8 = [&]() { v8f z;
#pragma unroll
    for (int r = 0; r < 8; ++r) z[r] = 0.f; return z; };

  // ---- GEMM1 / GEMM2: 256->256, in-place in xbuf[:,256:], A preloaded -----
#pragma unroll 1
  for (int layer = 0; layer < 2; ++layer) {
    const unsigned int off = layer ? OFF_WU : OFF_WM;
    const float* bb = layer ? bu : bm;
    const float  bs = layer ? 1.0f : 8.0f;     // k*bm folded into first bias
    ABFrag aS[2][8];
#pragma unroll
    for (int mt = 0; mt < 2; ++mt)
#pragma unroll
      for (int kt = 0; kt < 8; ++kt)
        aS[mt][kt] = loadA(xbuf + DDIM, XW, kt, m0a + 16 * mt);
#pragma unroll 1
    for (int nt = 0; nt < 16; ++nt) {
      v8f acc0 = zero8(), acc1 = zero8();
#pragma unroll
      for (int kt = 0; kt < 8; ++kt) {
        v16bf B = loadB(off, 8, nt, kt);
        acc0 = __builtin_amdgcn_wmma_f32_16x16x32_bf16(
            false, aS[0][kt].v, false, B, (short)0, acc0, false, false);
        acc1 = __builtin_amdgcn_wmma_f32_16x16x32_bf16(
            false, aS[1][kt].v, false, B, (short)0, acc1, false, false);
      }
      storeD(xbuf + DDIM, XW, nt, acc0, bb, bs, false, DDIM, m0a);
      storeD(xbuf + DDIM, XW, nt, acc1, bb, bs, false, DDIM, m0b);
    }
  }

  // ---- GEMM3: h = relu(x @ W0^T + b0)   K=512 (16 kt), N=160 (10 nt) ------
  // A streamed from LDS (too large to preload); B loaded once per (nt,kt).
#pragma unroll 1
  for (int nt = 0; nt < 10; ++nt) {
    v8f acc0 = zero8(), acc1 = zero8();
#pragma unroll
    for (int kt = 0; kt < 16; ++kt) {
      v16bf B = loadB(OFF_W0, 16, nt, kt);
      ABFrag a0 = loadA(xbuf, XW, kt, m0a);
      ABFrag a1 = loadA(xbuf, XW, kt, m0b);
      acc0 = __builtin_amdgcn_wmma_f32_16x16x32_bf16(
          false, a0.v, false, B, (short)0, acc0, false, false);
      acc1 = __builtin_amdgcn_wmma_f32_16x16x32_bf16(
          false, a1.v, false, B, (short)0, acc1, false, false);
    }
    storeD(hbuf, HW, nt, acc0, b0, 1.0f, true, HDIM, m0a);
    storeD(hbuf, HW, nt, acc1, b0, 1.0f, true, HDIM, m0b);
  }

  // ---- GEMM4 / GEMM5: h = relu(h @ Wi^T + bi)  K=160 (5 kt), N=160 (10 nt)
#pragma unroll 1
  for (int layer = 0; layer < 2; ++layer) {
    const unsigned int off = layer ? OFF_W2 : OFF_W1;
    const float* bb = layer ? b2 : b1;
    ABFrag aH[2][5];
#pragma unroll
    for (int mt = 0; mt < 2; ++mt)
#pragma unroll
      for (int kt = 0; kt < 5; ++kt)
        aH[mt][kt] = loadA(hbuf, HW, kt, m0a + 16 * mt);
#pragma unroll 1
    for (int nt = 0; nt < 10; ++nt) {
      v8f acc0 = zero8(), acc1 = zero8();
#pragma unroll
      for (int kt = 0; kt < 5; ++kt) {
        v16bf B = loadB(off, 5, nt, kt);
        acc0 = __builtin_amdgcn_wmma_f32_16x16x32_bf16(
            false, aH[0][kt].v, false, B, (short)0, acc0, false, false);
        acc1 = __builtin_amdgcn_wmma_f32_16x16x32_bf16(
            false, aH[1][kt].v, false, B, (short)0, acc1, false, false);
      }
      storeD(hbuf, HW, nt, acc0, bb, 1.0f, true, HDIM, m0a);
      storeD(hbuf, HW, nt, acc1, bb, 1.0f, true, HDIM, m0b);
    }
  }

  // ---- GEMM6: out = h @ W3^T + b3   K=160 (5 kt), N=256 (16 nt) -> HBM ----
  {
    ABFrag aH[2][5];
#pragma unroll
    for (int mt = 0; mt < 2; ++mt)
#pragma unroll
      for (int kt = 0; kt < 5; ++kt)
        aH[mt][kt] = loadA(hbuf, HW, kt, m0a + 16 * mt);
#pragma unroll 1
    for (int nt = 0; nt < 16; ++nt) {
      v8f acc0 = zero8(), acc1 = zero8();
#pragma unroll
      for (int kt = 0; kt < 5; ++kt) {
        v16bf B = loadB(OFF_W3, 5, nt, kt);
        acc0 = __builtin_amdgcn_wmma_f32_16x16x32_bf16(
            false, aH[0][kt].v, false, B, (short)0, acc0, false, false);
        acc1 = __builtin_amdgcn_wmma_f32_16x16x32_bf16(
            false, aH[1][kt].v, false, B, (short)0, acc1, false, false);
      }
      int col = nt * 16 + lm;
      float bb = b3[col];
#pragma unroll
      for (int r = 0; r < 8; ++r) {
        __builtin_nontemporal_store(
            acc0[r] + bb, &out[(size_t)(rowbase + m0a + r + 8 * lg) * DDIM + col]);
        __builtin_nontemporal_store(
            acc1[r] + bb, &out[(size_t)(rowbase + m0b + r + 8 * lg) * DDIM + col]);
      }
    }
  }
}

// ---------------------------------------------------------------------------
extern "C" void kernel_launch(void* const* d_in, const int* in_sizes, int n_in,
                              void* d_out, int out_size, void* d_ws, size_t ws_size,
                              hipStream_t stream) {
  const float* signal = (const float*)d_in[0];
  const float* comps  = (const float*)d_in[1];
  const float* Wm = (const float*)d_in[2];  const float* bm = (const float*)d_in[3];
  const float* Wu = (const float*)d_in[4];  const float* bu = (const float*)d_in[5];
  const float* W0 = (const float*)d_in[6];  const float* b0 = (const float*)d_in[7];
  const float* W1 = (const float*)d_in[8];  const float* b1 = (const float*)d_in[9];
  const float* W2 = (const float*)d_in[10]; const float* b2 = (const float*)d_in[11];
  const float* W3 = (const float*)d_in[12]; const float* b3 = (const float*)d_in[13];
  float* out = (float*)d_out;
  unsigned short* ws = (unsigned short*)d_ws;

  auto prep = [&](const float* W, unsigned int off, int oD, int iD, int NT, int KT) {
    int total = NT * KT * 256;
    prep_w<<<(total + 255) / 256, 256, 0, stream>>>(W, (unsigned int*)(ws + off), oD, iD, NT, KT);
  };
  prep(Wm, OFF_WM, 256, 256, 16, 8);
  prep(Wu, OFF_WU, 256, 256, 16, 8);
  prep(W0, OFF_W0, HDIM, EDIM + DDIM, 10, 16);
  prep(W1, OFF_W1, HDIM, HDIM, 10, 5);
  prep(W2, OFF_W2, HDIM, HDIM, 10, 5);
  prep(W3, OFF_W3, DDIM, HDIM, 16, 5);

  fused_main<<<NPTS / ROWS, THREADS, 0, stream>>>(signal, comps, bm, bu, b0, b1, b2, b3,
                                                  ws, out);
}
